// QVKAttentionLegacy_37503654429266
// MI455X (gfx1250) — matrix operand order repaired and checked by reading, not compiled, not run on hardware
//
#include <hip/hip_runtime.h>
#include <hip/hip_bf16.h>

typedef __attribute__((ext_vector_type(16))) _Float16 v16h;
typedef __attribute__((ext_vector_type(8)))  _Float16 v8h;
typedef __attribute__((ext_vector_type(4)))  _Float16 v4h;
typedef __attribute__((ext_vector_type(8)))  float    v8f;
typedef __attribute__((ext_vector_type(4)))  float    v4f;
typedef __attribute__((ext_vector_type(4)))  unsigned int u32x4;
typedef __attribute__((ext_vector_type(8)))  int      i32x8;
typedef __attribute__((ext_vector_type(4)))  int      i32x4;

#define LEN    2048
#define CH     64
#define NBH    64
#define STILE  64
#define TTILE  64
#define NCHUNK (LEN / STILE)

static constexpr size_t PLANE  = (size_t)NBH * LEN * CH;
static constexpr float  KSCALE = 0.35355339059327373f;                        // 64^-0.25
static constexpr float  QSCALE = 0.35355339059327373f * 1.4426950408889634f; // * log2(e)

#if __has_builtin(__builtin_amdgcn_tensor_load_to_lds)
#define HAVE_TDM 1
#else
#define HAVE_TDM 0
#endif

union V16U { v16h v; v8h h[2]; _Float16 e[16]; };

__device__ __forceinline__ float fast_exp2(float x) {
#if __has_builtin(__builtin_amdgcn_exp2f)
    return __builtin_amdgcn_exp2f(x);
#else
    return exp2f(x);
#endif
}
__device__ __forceinline__ void wait_tensorcnt_0() {
#if __has_builtin(__builtin_amdgcn_s_wait_tensorcnt)
    __builtin_amdgcn_s_wait_tensorcnt((short)0);
#else
    asm volatile("s_wait_tensorcnt 0x0" ::: "memory");
#endif
}
__device__ __forceinline__ void wait_tensorcnt_2() {
#if __has_builtin(__builtin_amdgcn_s_wait_tensorcnt)
    __builtin_amdgcn_s_wait_tensorcnt((short)2);
#else
    asm volatile("s_wait_tensorcnt 0x2" ::: "memory");
#endif
}

// ---- One-shot pre-pack: f32 qkv -> f16 {Q^T*qs, K^T*ks, V} in workspace ----
__global__ __launch_bounds__(256)
void pack_qkv(const float* __restrict__ qkv, _Float16* __restrict__ QT,
              _Float16* __restrict__ KT, _Float16* __restrict__ V16) {
    const int tid = threadIdx.x;
    const int s0  = blockIdx.x * STILE;
    const int bh  = blockIdx.y;
    const float* __restrict__ Qg = qkv + (size_t)bh * 3 * CH * LEN;
    const float* __restrict__ Kg = Qg + (size_t)CH * LEN;
    const float* __restrict__ Vg = Qg + (size_t)2 * CH * LEN;

    const int sL = tid & 63;
    const int cB = (tid >> 6) << 4;
    #pragma unroll
    for (int j = 0; j < 2; ++j) {
        v8h hq, hk;
        #pragma unroll
        for (int c = 0; c < 8; ++c) {
            const int ch = cB + j * 8 + c;
            hq[c] = (_Float16)(Qg[(size_t)ch * LEN + s0 + sL] * QSCALE);
            hk[c] = (_Float16)(Kg[(size_t)ch * LEN + s0 + sL] * KSCALE);
        }
        const size_t row = ((size_t)bh * LEN + s0 + sL) * CH + cB + j * 8;
        *(v8h*)(QT + row) = hq;
        *(v8h*)(KT + row) = hk;
    }
    #pragma unroll
    for (int i = 0; i < 4; ++i) {
        const int idx = tid + i * 256;
        const int c   = idx >> 4;
        const int s4  = (idx & 15) << 2;
        const v4f vv  = *(const v4f*)&Vg[(size_t)c * LEN + s0 + s4];
        v4h hv;
        hv[0] = (_Float16)vv[0]; hv[1] = (_Float16)vv[1];
        hv[2] = (_Float16)vv[2]; hv[3] = (_Float16)vv[3];
        *(v4h*)&V16[((size_t)bh * CH + c) * LEN + s0 + s4] = hv;
    }
}

// ---- Shared per-chunk compute core (transposed: S^T = K^T Q, O^T = V P^T) --
// C-layout puts s along VGPRs and t along lanes -> softmax reductions are
// in-lane VALU + ONE shfl_xor(16); no bpermute trees, no row-sum WMMAs.
__device__ __forceinline__ void attn_chunk_t(
    const _Float16 (*__restrict__ ldsKT)[CH], const _Float16 (*__restrict__ ldsV)[STILE],
    _Float16 (*__restrict__ ldsPT)[STILE],
    const v16h& bq0, const v16h& bq1,
    int ln16, int hb, int bsel, float& m, float& l, v8f o[4])
{
    // S^T(64s x 16t) = (K^T)(Q) : 8 WMMAs, A = K^T tile rows, B = Q cols
    v8f sfT[4];
    #pragma unroll
    for (int n = 0; n < 4; ++n) {
        const int sn = n * 16 + ln16;
        V16U ak0, ak1;
        ak0.h[0] = *(const v8h*)&ldsKT[sn][bsel];
        ak0.h[1] = *(const v8h*)&ldsKT[sn][8 + bsel];
        ak1.h[0] = *(const v8h*)&ldsKT[sn][32 + bsel];
        ak1.h[1] = *(const v8h*)&ldsKT[sn][40 + bsel];
        v8f acc = {};
        acc = __builtin_amdgcn_wmma_f32_16x16x32_f16(false, ak0.v, false, bq0, (short)0, acc, false, false);
        acc = __builtin_amdgcn_wmma_f32_16x16x32_f16(false, ak1.v, false, bq1, (short)0, acc, false, false);
        sfT[n] = acc;
    }

    // Per-lane max over the 32 in-lane s values, then one cross-half combine
    float mx = sfT[0][0];
    #pragma unroll
    for (int n = 0; n < 4; ++n)
        #pragma unroll
        for (int r = 0; r < 8; ++r) mx = fmaxf(mx, sfT[n][r]);
    mx = fmaxf(mx, __shfl_xor(mx, 16, 32));

    const float mn    = fmaxf(m, mx);
    const float alpha = fast_exp2(m - mn);
    m = mn;

    float rs = 0.0f;
    #pragma unroll
    for (int n = 0; n < 4; ++n)
        #pragma unroll
        for (int r = 0; r < 8; ++r) {
            const float p = fast_exp2(sfT[n][r] - mn);
            sfT[n][r] = p;
            rs += p;
        }
    rs += __shfl_xor(rs, 16, 32);
    l = l * alpha + rs;
    #pragma unroll
    for (int ct = 0; ct < 4; ++ct) o[ct] = o[ct] * alpha;

    // Stage P^T[t][s] (C-layout -> B-layout via wave-private LDS)
    #pragma unroll
    for (int n = 0; n < 4; ++n)
        #pragma unroll
        for (int r = 0; r < 8; ++r)
            ldsPT[ln16][n * 16 + hb + r] = (_Float16)sfT[n][r];
    asm volatile("" ::: "memory");  // same-wave DS ops are in-order (ISA 7.3)

    const v16h bp0 = *(const v16h*)&ldsPT[ln16][bsel];       // s 0..31
    const v16h bp1 = *(const v16h*)&ldsPT[ln16][32 + bsel];  // s 32..63

    // O^T(64c x 16t) += V P^T : 8 WMMAs, A = V tile rows, B = P^T cols
    #pragma unroll
    for (int ct = 0; ct < 4; ++ct) {
        const int c = ct * 16 + ln16;
        V16U av0, av1;
        av0.h[0] = *(const v8h*)&ldsV[c][bsel];
        av0.h[1] = *(const v8h*)&ldsV[c][8 + bsel];
        av1.h[0] = *(const v8h*)&ldsV[c][32 + bsel];
        av1.h[1] = *(const v8h*)&ldsV[c][40 + bsel];
        o[ct] = __builtin_amdgcn_wmma_f32_16x16x32_f16(false, av0.v, false, bp0, (short)0, o[ct], false, false);
        o[ct] = __builtin_amdgcn_wmma_f32_16x16x32_f16(false, av1.v, false, bp1, (short)0, o[ct], false, false);
    }
}

__device__ __forceinline__ void attn_epilogue_t(
    float (*__restrict__ ldsO)[TTILE], float* __restrict__ Ob,
    int tid, int ln16, int hb, int wave, const v8f o[4], float l)
{
    __syncthreads();
    const float inv = 1.0f / l;
    #pragma unroll
    for (int ct = 0; ct < 4; ++ct) {
        #pragma unroll
        for (int r = 0; r < 8; ++r)          // O^T C-layout: M=c in VGPRs, N=t in lanes
            ldsO[ct * 16 + hb + r][wave * 16 + ln16] = o[ct][r] * inv;
    }
    __syncthreads();
    #pragma unroll
    for (int i = 0; i < 8; ++i) {
        const int idx = tid + i * 128;
        const int c   = idx >> 4;
        const int t4  = (idx & 15) << 2;
        *(v4f*)&Ob[(size_t)c * LEN + t4] = *(const v4f*)&ldsO[c][t4];
    }
}

#if HAVE_TDM
// ---- TDM descriptor helpers (ISA 8.3/8.4 bit layout) -----------------------
__device__ __forceinline__ void tdm_issue(unsigned lds_addr, unsigned long long gaddr,
                                          const i32x8& g1) {
    u32x4 g0;
    g0[0] = 1u;                                             // count=1, user mode
    g0[1] = lds_addr;                                       // lds_addr[31:0]
    g0[2] = (unsigned)gaddr;                                // global_addr[31:0]
    g0[3] = (unsigned)((gaddr >> 32) & 0x01FFFFFFull) | 0x80000000u; // addr[56:32] | type=2
    i32x4 z4 = {};
    i32x8 z8 = {};
    __builtin_amdgcn_tensor_load_to_lds(g0, g1, z4, z4, z8, 0);
}
__device__ __forceinline__ i32x8 tdm_group1(int tile0, int tile1, int stride0) {
    i32x8 g;
    g[0] = 0x00030000;                       // data_size=3 (8B)
    g[1] = (int)0xFFFF0000;                  // tensor_dim0 = 0x7FFFFFFF (no OOB clip)
    g[2] = (int)0xFFFF7FFF;
    g[3] = 0x00007FFF | (tile0 << 16);       // tile_dim0
    g[4] = tile1;                            // tile_dim1 (0 = 1D)
    g[5] = stride0;                          // tensor_dim0_stride (8B units)
    g[6] = 0;
    g[7] = 0;
    return g;
}

// ---- Main kernel: double-buffered TDM staging ------------------------------
__global__ __launch_bounds__(128)
void attn_flash_tdm(const _Float16* __restrict__ QT, const _Float16* __restrict__ KT,
                    const _Float16* __restrict__ V16, float* __restrict__ out) {
    __shared__ __attribute__((aligned(128))) unsigned char smem[40 * 1024];

    const int tid  = threadIdx.x;
    const int lane = tid & 31;
    const int wave = tid >> 5;
    const int bh   = blockIdx.y;
    const int tBlk = blockIdx.x * TTILE;

    const int ln16 = lane & 15;
    const int hb   = (lane & 16) >> 1;
    const int bsel = lane & 16;

    unsigned char* sm = smem;
    const unsigned smemOff = (unsigned)(unsigned long long)(uintptr_t)sm;
    _Float16 (*ldsPT)[STILE] = (_Float16 (*)[STILE])(sm + 32768 + wave * 2048);
    float    (*ldsO)[TTILE]  = (float (*)[TTILE])sm;

    const unsigned long long gK = (unsigned long long)(KT + (size_t)bh * LEN * CH);
    const unsigned long long gV = (unsigned long long)(V16 + (size_t)bh * CH * LEN);
    const i32x8 g1K = tdm_group1(1024, 0, 0);    // 8KB contiguous K^T tile
    const i32x8 g1V = tdm_group1(16, 64, 512);   // 64 rows x 128B, 4KB row stride

    // Persistent Q B-fragments (lane = t, halves = channel)
    const int tq = tBlk + wave * 16 + ln16;
    const _Float16* qrow = QT + ((size_t)bh * LEN + tq) * CH;
    const v16h bq0 = *(const v16h*)(qrow + bsel);        // c 0..31
    const v16h bq1 = *(const v16h*)(qrow + 32 + bsel);   // c 32..63

    float m = -1.0e30f, l = 0.0f;
    v8f o[4];
    #pragma unroll
    for (int ct = 0; ct < 4; ++ct) o[ct] = {};

    if (wave == 0) {                         // prefetch chunk 0 into buffer 0
        tdm_issue(smemOff, gK, g1K);
        tdm_issue(smemOff + 8192u, gV, g1V);
    }

    for (int i = 0; i < NCHUNK; ++i) {
        const unsigned curOff = (unsigned)(i & 1) * 16384u;
        const unsigned nxtOff = curOff ^ 16384u;
        __syncthreads();                     // buf[nxt] free for TDM overwrite
        if (wave == 0) {
            if (i + 1 < NCHUNK) {            // stream next chunk while computing
                const size_t off = (size_t)(i + 1) * STILE;
                tdm_issue(smemOff + nxtOff, gK + off * CH * 2, g1K);
                tdm_issue(smemOff + nxtOff + 8192u, gV + off * 2, g1V);
                wait_tensorcnt_2();          // pair i landed (in-order completion)
            } else {
                wait_tensorcnt_0();
            }
        }
        __syncthreads();                     // publish buf[cur] to all waves
        attn_chunk_t((const _Float16 (*)[CH])(sm + curOff),
                     (const _Float16 (*)[STILE])(sm + curOff + 8192u),
                     ldsPT, bq0, bq1, ln16, hb, bsel, m, l, o);
    }

    attn_epilogue_t(ldsO, out + (size_t)bh * CH * LEN + tBlk,
                    tid, ln16, hb, wave, o, l);
}
#endif // HAVE_TDM

// ---- Fallback kernels (no TDM, and/or no workspace) ------------------------
template<bool PP>
__global__ __launch_bounds__(128)
void attn_flash_wmma(const float* __restrict__ qkv,
                     const _Float16* __restrict__ QT,
                     const _Float16* __restrict__ KT,
                     const _Float16* __restrict__ V16,
                     float* __restrict__ out) {
    __shared__ __attribute__((aligned(128))) unsigned char smem[24 * 1024];

    const int tid  = threadIdx.x;
    const int lane = tid & 31;
    const int wave = tid >> 5;
    const int bh   = blockIdx.y;
    const int tBlk = blockIdx.x * TTILE;

    const int ln16 = lane & 15;
    const int hb   = (lane & 16) >> 1;
    const int bsel = lane & 16;

    _Float16 (*ldsKT)[CH]    = (_Float16 (*)[CH])smem;
    _Float16 (*ldsV)[STILE]  = (_Float16 (*)[STILE])(smem + 8192);
    _Float16 (*ldsPT)[STILE] = (_Float16 (*)[STILE])(smem + 16384 + wave * 2048);
    float    (*ldsO)[TTILE]  = (float (*)[TTILE])smem;

    const float* __restrict__ Qg = qkv + (size_t)bh * 3 * CH * LEN;
    const float* __restrict__ Kg = Qg + (size_t)CH * LEN;
    const float* __restrict__ Vg = Qg + (size_t)2 * CH * LEN;

    const int tq = tBlk + wave * 16 + ln16;
    v16h bq0, bq1;
    if constexpr (PP) {
        const _Float16* qrow = QT + ((size_t)bh * LEN + tq) * CH;
        bq0 = *(const v16h*)(qrow + bsel);
        bq1 = *(const v16h*)(qrow + 32 + bsel);
    } else {
        #pragma unroll
        for (int i = 0; i < 16; ++i) {       // B-layout: element i <-> channel bsel+i
            bq0[i] = (_Float16)(Qg[(size_t)(bsel + i) * LEN + tq] * QSCALE);
            bq1[i] = (_Float16)(Qg[(size_t)(32 + bsel + i) * LEN + tq] * QSCALE);
        }
    }

    float m = -1.0e30f, l = 0.0f;
    v8f o[4];
    #pragma unroll
    for (int ct = 0; ct < 4; ++ct) o[ct] = {};

    const int sL = tid & 63;
    const int cB = (tid >> 6) << 5;

    for (int s0 = 0; s0 < LEN; s0 += STILE) {
        __syncthreads();
        if constexpr (PP) {
            const _Float16* ksrc = KT + ((size_t)bh * LEN + s0) * CH;
            _Float16* kdst = &ldsKT[0][0];
            #pragma unroll
            for (int i = 0; i < 4; ++i) {
                const int e = (tid + i * 128) * 8;
                *(v8h*)(kdst + e) = *(const v8h*)(ksrc + e);
            }
            #pragma unroll
            for (int i = 0; i < 4; ++i) {
                const int idx = tid + i * 128;
                const int c = idx >> 3, j = (idx & 7) << 3;
                *(v8h*)&ldsV[c][j] =
                    *(const v8h*)&V16[((size_t)bh * CH + c) * LEN + s0 + j];
            }
        } else {
            #pragma unroll
            for (int j = 0; j < 4; ++j) {
                v8h hv;
                #pragma unroll
                for (int c = 0; c < 8; ++c)
                    hv[c] = (_Float16)(Kg[(size_t)(cB + j * 8 + c) * LEN + s0 + sL] * KSCALE);
                *(v8h*)&ldsKT[sL][cB + j * 8] = hv;
            }
            #pragma unroll
            for (int i = 0; i < 8; ++i) {
                const int idx = tid + i * 128;
                const int c   = idx >> 4;
                const int s4  = (idx & 15) << 2;
                const v4f vv  = *(const v4f*)&Vg[(size_t)c * LEN + s0 + s4];
                v4h hv;
                hv[0] = (_Float16)vv[0]; hv[1] = (_Float16)vv[1];
                hv[2] = (_Float16)vv[2]; hv[3] = (_Float16)vv[3];
                *(v4h*)&ldsV[c][s4] = hv;
            }
        }
        __syncthreads();
        attn_chunk_t(ldsKT, ldsV, ldsPT, bq0, bq1, ln16, hb, bsel, m, l, o);
    }

    attn_epilogue_t(ldsO, out + (size_t)bh * CH * LEN + tBlk,
                    tid, ln16, hb, wave, o, l);
}

extern "C" void kernel_launch(void* const* d_in, const int* in_sizes, int n_in,
                              void* d_out, int out_size, void* d_ws, size_t ws_size,
                              hipStream_t stream) {
    const float* qkv = (const float*)d_in[0];
    float* out = (float*)d_out;
    dim3 grid(LEN / TTILE, NBH);
    const size_t need = 3 * PLANE * sizeof(_Float16);   // ~50.3 MB
    if (d_ws != nullptr && ws_size >= need) {
        _Float16* QT  = (_Float16*)d_ws;
        _Float16* KT  = QT + PLANE;
        _Float16* V16 = KT + PLANE;
        pack_qkv<<<dim3(LEN / STILE, NBH), 256, 0, stream>>>(qkv, QT, KT, V16);
#if HAVE_TDM
        attn_flash_tdm<<<grid, 128, 0, stream>>>(QT, KT, V16, out);
#else
        attn_flash_wmma<true><<<grid, 128, 0, stream>>>(qkv, QT, KT, V16, out);
#endif
    } else {
        attn_flash_wmma<false><<<grid, 128, 0, stream>>>(qkv, nullptr, nullptr, nullptr, out);
    }
}
